// SiameseLSTM_11536282157577
// MI455X (gfx1250) — compile-verified, matrix operand order checked by reading, not run on hardware
//
#include <hip/hip_runtime.h>
#include <hip/hip_bf16.h>

typedef __attribute__((ext_vector_type(16))) _Float16 v16h;
typedef __attribute__((ext_vector_type(8)))  float    v8f;

#define TT   128
#define BB   64
#define DD   300
#define HH   256
#define G4H  1024
#define KPAD 320
#define KT1  10            // k-tiles for D=300 padded to 320
#define KT2  8             // k-tiles for H=256
#define NT   64            // n-tiles over 4H=1024
#define MT_TOT 512         // (T*B)/16 m-tiles per sequence

// CDNA5 16-bit A/B fragment: element e of lane `lane` holds K-offset kl within
// a 16x32 tile (ISA 7.12.2): lanes 0-15 cover (K&15)<8, lanes 16-31 the rest.
__device__ __forceinline__ int kl_of(int e, int lane) {
  return (((e >> 1) & 3) << 1) + (e & 1) + ((e & 8) ? 16 : 0) + ((lane & 16) ? 8 : 0);
}

// ---------------------------------------------------------------------------
// Pack W_ih (f32 [1024,300]) and W_hh (f32 [1024,256]) into f16 WMMA B-frags:
// B[k][n] = W[n][k]; storage [ntile][ktile][lane][16 halves] (1KB per frag).
// ---------------------------------------------------------------------------
__global__ void pack_weights(const float* __restrict__ W_ih,
                             const float* __restrict__ W_hh,
                             _Float16* __restrict__ B1,
                             _Float16* __restrict__ B2) {
  const int n1 = NT * KT1 * 512;
  const int n2 = NT * KT2 * 512;
  for (int idx = blockIdx.x * blockDim.x + threadIdx.x; idx < n1 + n2;
       idx += gridDim.x * blockDim.x) {
    if (idx < n1) {
      int frag = idx >> 9, w = idx & 511;
      int lane = w >> 4, e = w & 15;
      int ntile = frag / KT1, kt = frag % KT1;
      int n = ntile * 16 + (lane & 15);
      int k = kt * 32 + kl_of(e, lane);
      float v = (k < DD) ? W_ih[n * DD + k] : 0.f;
      B1[idx] = (_Float16)v;
    } else {
      int j = idx - n1;
      int frag = j >> 9, w = j & 511;
      int lane = w >> 4, e = w & 15;
      int ntile = frag / KT2, kt = frag % KT2;
      int n = ntile * 16 + (lane & 15);
      int k = kt * 32 + kl_of(e, lane);
      B2[j] = (_Float16)W_hh[n * HH + k];
    }
  }
}

// ---------------------------------------------------------------------------
// gx = emb[s] @ W_ih^T + b_ih + b_hh, stored as f32 C-fragments:
// gxf[seq][mtile][ntile][lane][8].  One wave per 16-row M-tile.
// ---------------------------------------------------------------------------
__global__ void __launch_bounds__(32) gx_gemm(const int* __restrict__ s1,
                                              const int* __restrict__ s2,
                                              const float* __restrict__ emb,
                                              const _Float16* __restrict__ B1,
                                              const float* __restrict__ b_ih,
                                              const float* __restrict__ b_hh,
                                              float* __restrict__ gxf) {
  __shared__ __align__(32) _Float16 xl[16][KPAD];   // 10KB
  const int lane  = threadIdx.x;
  const int seq   = blockIdx.x >> 9;
  const int mtile = blockIdx.x & 511;
  const int* s = seq ? s2 : s1;

  // Gather 16 embedding rows (f32 -> f16, zero-pad K to 320)
  for (int r = 0; r < 16; ++r) {
    int m = mtile * 16 + r;
    int t = m >> 6, b = m & 63;          // B == 64
    long long row = s[t * BB + b];
    const float* src = emb + row * DD;
    for (int k = lane; k < KPAD; k += 32)
      xl[r][k] = (k < DD) ? (_Float16)src[k] : (_Float16)0.f;
  }
  __syncthreads();

  // Build the 10 A-fragments once (kept in VGPRs for all 64 n-tiles)
  v16h a[KT1];
  const int row = lane & 15;
#pragma unroll
  for (int kt = 0; kt < KT1; ++kt) {
#pragma unroll
    for (int e = 0; e < 16; ++e)
      a[kt][e] = xl[row][kt * 32 + kl_of(e, lane)];
  }

  for (int ntile = 0; ntile < NT; ++ntile) {
    v8f acc = {};
#pragma unroll
    for (int kt = 0; kt < KT1; ++kt) {
      v16h bf = *(const v16h*)(B1 + (((ntile * KT1 + kt) << 9) + (lane << 4)));
      acc = __builtin_amdgcn_wmma_f32_16x16x32_f16(false, a[kt], false, bf,
                                                   (short)0, acc, false, false);
    }
    int n = ntile * 16 + (lane & 15);
    float bias = b_ih[n] + b_hh[n];
#pragma unroll
    for (int r = 0; r < 8; ++r) acc[r] += bias;
    size_t off = ((((size_t)seq * MT_TOT + mtile) * NT + ntile) << 8) + (lane << 3);
    *(v8f*)(gxf + off) = acc;
  }
}

// ---------------------------------------------------------------------------
// Persistent recurrence: one 1024-thread workgroup (32 waves) per LSTM.
// Wave (mtile = w>>3, wslot = w&7) owns h-tiles {wslot, wslot+8} of its M-tile.
// h lives in LDS already in A-fragment layout; c stays in VGPRs for 128 steps.
// ---------------------------------------------------------------------------
__global__ void __launch_bounds__(1024) lstm_scan(const float* __restrict__ gxf,
                                                  const _Float16* __restrict__ B2,
                                                  const float* __restrict__ h0a,
                                                  const float* __restrict__ c0a,
                                                  const float* __restrict__ h0b,
                                                  const float* __restrict__ c0b,
                                                  float* __restrict__ hfin) {
  __shared__ __align__(32) _Float16 Af[4][KT2][32][16];   // 32KB: h as A-frags
  const int seq = blockIdx.x;
  const float* h0 = seq ? h0b : h0a;
  const float* c0 = seq ? c0b : c0a;
  const int tid = threadIdx.x;
  const int lane = tid & 31, wave = tid >> 5;
  const int mtile = wave >> 3, wslot = wave & 7;

  // h0 -> LDS A-fragment layout
  for (int idx = tid; idx < 4 * KT2 * 32 * 16; idx += 1024) {
    int e = idx & 15, ln = (idx >> 4) & 31, kt = (idx >> 9) & 7, mt = idx >> 12;
    int m = mt * 16 + (ln & 15);
    int k = kt * 32 + kl_of(e, ln);
    Af[mt][kt][ln][e] = (_Float16)h0[m * HH + k];
  }

  // c0 -> registers in C/D fragment layout
  float cf[2][8], hf[2][8];
#pragma unroll
  for (int ji = 0; ji < 2; ++ji) {
    int jt = wslot + ji * 8;
#pragma unroll
    for (int r = 0; r < 8; ++r) {
      int m = mtile * 16 + r + ((lane & 16) ? 8 : 0);
      int j = jt * 16 + (lane & 15);
      cf[ji][r] = c0[m * HH + j];
      hf[ji][r] = 0.f;
    }
  }

  for (int t = 0; t < TT; ++t) {
    __syncthreads();   // A-frags (h_{t-1}) visible
#pragma unroll
    for (int ji = 0; ji < 2; ++ji) {
      int jt = wslot + ji * 8;
      v8f acc[4];
#pragma unroll
      for (int g = 0; g < 4; ++g) {       // gate order i,f,g,o
        int ntile = g * 16 + jt;
        size_t off = ((((size_t)seq * MT_TOT + (t * 4 + mtile)) * NT + ntile) << 8)
                     + (lane << 3);
        acc[g] = *(const v8f*)(gxf + off);
#pragma unroll
        for (int kt = 0; kt < KT2; ++kt) {
          v16h a = *(const v16h*)&Af[mtile][kt][lane][0];
          v16h b = *(const v16h*)(B2 + (((ntile * KT2 + kt) << 9) + (lane << 4)));
          acc[g] = __builtin_amdgcn_wmma_f32_16x16x32_f16(false, a, false, b,
                                                          (short)0, acc[g],
                                                          false, false);
        }
      }
#pragma unroll
      for (int r = 0; r < 8; ++r) {
        float ig = 1.f / (1.f + __expf(-acc[0][r]));
        float fg = 1.f / (1.f + __expf(-acc[1][r]));
        float gg = tanhf(acc[2][r]);
        float og = 1.f / (1.f + __expf(-acc[3][r]));
        float c = fg * cf[ji][r] + ig * gg;
        cf[ji][r] = c;
        hf[ji][r] = og * tanhf(c);
      }
    }
    __syncthreads();   // everyone done reading old A-frags

    // Scatter h_t into LDS A-fragment layout for step t+1
#pragma unroll
    for (int ji = 0; ji < 2; ++ji) {
      int jt = wslot + ji * 8;
#pragma unroll
      for (int r = 0; r < 8; ++r) {
        int mrow = r + ((lane & 16) ? 8 : 0);
        int j = jt * 16 + (lane & 15);
        int kt = j >> 5, klv = j & 31;
        int ln2 = mrow + ((klv & 8) ? 16 : 0);
        int e = 2 * (((klv & 7) >> 1) + ((klv & 16) ? 4 : 0)) + (klv & 1);
        Af[mtile][kt][ln2][e] = (_Float16)hf[ji][r];
      }
    }

    // Prefetch next step's gx fragments into cache (global_prefetch_b8)
    if (t + 1 < TT) {
      size_t poff = ((((size_t)seq * MT_TOT + ((t + 1) * 4 + mtile)) * NT + wslot) << 8)
                    + (lane << 3);
      __builtin_prefetch(gxf + poff, 0, 0);
    }
  }

  // Final hidden state -> global
#pragma unroll
  for (int ji = 0; ji < 2; ++ji) {
    int jt = wslot + ji * 8;
#pragma unroll
    for (int r = 0; r < 8; ++r) {
      int m = mtile * 16 + r + ((lane & 16) ? 8 : 0);
      int j = jt * 16 + (lane & 15);
      hfin[(seq * BB + m) * HH + j] = hf[ji][r];
    }
  }
}

// ---------------------------------------------------------------------------
// y[b] = clip(exp(-||ha[b]-hb[b]||_1), 1e-7, 1-1e-7)
// ---------------------------------------------------------------------------
__global__ void l1_exp(const float* __restrict__ hfin, float* __restrict__ out) {
  int b = threadIdx.x;
  if (b < BB) {
    float s = 0.f;
    for (int j = 0; j < HH; ++j)
      s += fabsf(hfin[b * HH + j] - hfin[(BB + b) * HH + j]);
    float y = __expf(-s);
    y = fminf(fmaxf(y, 1e-7f), 1.f - 1e-7f);
    out[b] = y;
  }
}

extern "C" void kernel_launch(void* const* d_in, const int* in_sizes, int n_in,
                              void* d_out, int out_size, void* d_ws, size_t ws_size,
                              hipStream_t stream) {
  const int*   s1   = (const int*)d_in[0];
  const int*   s2   = (const int*)d_in[1];
  const float* emb  = (const float*)d_in[2];
  const float* W_ih = (const float*)d_in[3];
  const float* W_hh = (const float*)d_in[4];
  const float* b_ih = (const float*)d_in[5];
  const float* b_hh = (const float*)d_in[6];
  const float* h0a  = (const float*)d_in[7];
  const float* c0a  = (const float*)d_in[8];
  const float* h0b  = (const float*)d_in[9];
  const float* c0b  = (const float*)d_in[10];

  char* ws = (char*)d_ws;
  _Float16* B1   = (_Float16*)ws;                       // 655,360 B
  _Float16* B2   = (_Float16*)(ws + 655360);            // 524,288 B
  float*    hfin = (float*)(ws + 655360 + 524288);      // 131,072 B
  float*    gxf  = (float*)(ws + 1310720);              // 64 MB of C-frags

  pack_weights<<<256, 256, 0, stream>>>(W_ih, W_hh, B1, B2);
  gx_gemm<<<1024, 32, 0, stream>>>(s1, s2, emb, B1, b_ih, b_hh, gxf);
  lstm_scan<<<2, 1024, 0, stream>>>(gxf, B2, h0a, c0a, h0b, c0b, hfin);
  l1_exp<<<1, 64, 0, stream>>>(hfin, (float*)d_out);
}
